// MSSSIMLoss_5617817223620
// MI455X (gfx1250) — matrix operand order repaired and checked by reading, not compiled, not run on hardware
//
#include <hip/hip_runtime.h>
#include <hip/hip_bf16.h>

// MS-SSIM loss for MI455X (gfx1250, wave32).
// Separable 11-tap Gaussian conv mapped onto v_wmma_f32_16x16x32_bf16 via a
// banded weight A-matrix; f32 accumulation for the variance cancellation.
// Wave-index-derived control flow goes through readfirstlane so WMMA always
// executes with EXEC all-ones (scalar loops, no EXEC masking around wmma).

typedef __bf16 bf16x16 __attribute__((ext_vector_type(16)));
typedef float  v8f     __attribute__((ext_vector_type(8)));

#define TILE   64          // output tile per workgroup
#define PAD    80          // staged input tile (74 used + alignment pad)
#define NMAPS  5           // x, y, x^2, y^2, x*y
#define HOUT   502         // 512 - 11 + 1
#define NB     16
#define NC     5
#define HW     512

__device__ __forceinline__ unsigned pack_bf16(float lo, float hi) {
  union { float f; unsigned u; } a, b; a.f = lo; b.f = hi;
  unsigned ulo = (a.u + 0x7FFFu + ((a.u >> 16) & 1u)) >> 16;
  unsigned uhi = (b.u + 0x7FFFu + ((b.u >> 16) & 1u)) >> 16;
  return (uhi << 16) | (ulo & 0xFFFFu);
}
__device__ __forceinline__ unsigned short f2bf(float f) {
  union { float f; unsigned u; } a; a.f = f;
  return (unsigned short)((a.u + 0x7FFFu + ((a.u >> 16) & 1u)) >> 16);
}
// banded weight lookup, fully unrolled selects so w[] stays in VGPRs
__device__ __forceinline__ float wband(int t, const float* w) {
  float r = 0.f;
#pragma unroll
  for (int i = 0; i < 11; i++) r = (t == i) ? w[i] : r;
  return r;
}

// ---------------- pass 0: scratch init ----------------
__global__ void k_init(unsigned int* ws) {
  if (threadIdx.x == 0) {
    ws[0] = 0xFFFFFFFFu;          // min key
    ws[1] = 0u;                   // max key
    ((float*)ws)[2] = 0.f;        // loss accumulator
  }
}

// ---------------- pass 1: global min/max of input ----------------
__global__ void k_minmax(const float* __restrict__ x, int n, unsigned int* __restrict__ ws) {
  unsigned lmin = 0xFFFFFFFFu, lmax = 0u;
  for (int i = blockIdx.x * blockDim.x + threadIdx.x; i < n; i += gridDim.x * blockDim.x) {
    union { float f; unsigned u; } c; c.f = x[i];
    unsigned key = (c.u & 0x80000000u) ? ~c.u : (c.u | 0x80000000u); // order-preserving
    lmin = key < lmin ? key : lmin;
    lmax = key > lmax ? key : lmax;
  }
#pragma unroll
  for (int off = 16; off > 0; off >>= 1) {
    unsigned omin = __shfl_down(lmin, (unsigned)off, 32);
    unsigned omax = __shfl_down(lmax, (unsigned)off, 32);
    lmin = omin < lmin ? omin : lmin;
    lmax = omax > lmax ? omax : lmax;
  }
  if ((threadIdx.x & 31) == 0) {
    atomicMin(ws + 0, lmin);
    atomicMax(ws + 1, lmax);
  }
}

// ---------------- pass 2: fused separable WMMA conv + SSIM + sum ----------------
__global__ void __launch_bounds__(512)
msssim_main(const float* __restrict__ input, const float* __restrict__ target,
            const float* __restrict__ filt, unsigned int* __restrict__ ws)
{
  extern __shared__ unsigned short smem[];
  unsigned short* s_map = smem;                         // [5][PAD*PAD] bf16 source maps
  unsigned short* s_ht  = smem + NMAPS * PAD * PAD;     // [PAD*PAD]  transposed H pass

  const int tid  = threadIdx.x;
  const int lane = tid & 31;
  // wave index as an SGPR: keeps all wmma-adjacent control flow scalar
  const int wvu  = __builtin_amdgcn_readfirstlane(tid >> 5);   // 0..15
  const int nrow = lane & 15;         // N index within 16x16 fragments
  const int hi   = lane >> 4;         // half-wave select

  const int b    = blockIdx.x >> 6;
  const int tile = blockIdx.x & 63;
  const int th0  = (tile >> 3) * TILE;
  const int tw0  = (tile & 7) * TILE;

  // dynamic-range constants from pass 1
  unsigned kmin = ws[0], kmax = ws[1];
  union { unsigned u; float f; } cv;
  cv.u = (kmax & 0x80000000u) ? (kmax & 0x7FFFFFFFu) : ~kmax; float fmaxv = cv.f;
  cv.u = (kmin & 0x80000000u) ? (kmin & 0x7FFFFFFFu) : ~kmin; float fminv = cv.f;
  float max_val = fmaxv > 128.f ? 255.f : 1.f;
  float min_val = fminv < -0.5f ? -1.f : 0.f;
  float L  = max_val - min_val;
  float C1 = (0.01f * L) * (0.01f * L);
  float C2 = (0.03f * L) * (0.03f * L);

  float prodcs[8], lterm[8];
#pragma unroll
  for (int v = 0; v < 8; v++) { prodcs[v] = 1.f; lterm[v] = 0.f; }

  for (int c = 0; c < NC; c++) {
    // 1-D taps from the 2-D separable filter: w1[i] = w2d[i][5] / sqrt(w2d[5][5])
    float w1c[11];
    {
      float inv = rsqrtf(filt[c * 121 + 60]);
#pragma unroll
      for (int t = 0; t < 11; t++) w1c[t] = filt[c * 121 + t * 11 + 5] * inv;
    }
    // banded A-matrix (16x32 bf16): A[m,k] = w1[k-m]; layout per ISA 16-bit A table
    unsigned a_pack[8];
#pragma unroll
    for (int j = 0; j < 8; j++) {
      int k0 = ((j >> 2) << 4) + ((j & 3) << 1) + (hi << 3);
      a_pack[j] = pack_bf16(wband(k0 - nrow, w1c), wband(k0 + 1 - nrow, w1c));
    }

    __syncthreads();
    // stage bf16 source maps (clamped halo loads keep band-waste lanes finite)
    const float* xin = input  + (size_t)(b * NC + c) * HW * HW;
    const float* yin = target + (size_t)(b * NC + c) * HW * HW;
    for (int idx = tid; idx < PAD * PAD; idx += 512) {
      int r = idx / PAD, q = idx - r * PAD;
      int gr = th0 + r; gr = gr < HW ? gr : HW - 1;
      int gc = tw0 + q; gc = gc < HW ? gc : HW - 1;
      float xv = xin[gr * HW + gc];
      float yv = yin[gr * HW + gc];
      if (c + 1 < NC) {  // overlap next channel's HBM latency with this channel's WMMAs
        __builtin_prefetch(&xin[HW * HW + gr * HW + gc], 0, 1);
        __builtin_prefetch(&yin[HW * HW + gr * HW + gc], 0, 1);
      }
      s_map[0 * PAD * PAD + idx] = f2bf(xv);
      s_map[1 * PAD * PAD + idx] = f2bf(yv);
      s_map[2 * PAD * PAD + idx] = f2bf(xv * xv);
      s_map[3 * PAD * PAD + idx] = f2bf(yv * yv);
      s_map[4 * PAD * PAD + idx] = f2bf(xv * yv);
    }
    __syncthreads();

    float stat[NMAPS][8];

#pragma unroll
    for (int m = 0; m < NMAPS; m++) {
      const unsigned short* src = s_map + m * PAD * PAD;

      // horizontal pass: 20 fragments of H (80 rows x 64 cols); scalar loop bound
      // (wvu is SGPR) -> uniform SCC branch, EXEC stays all-ones around wmma.
      for (int t = wvu; t < 20; t += 16) {
        int rbase = (t >> 2) * 16, cbase = (t & 3) * 16;
        union { unsigned u[8]; bf16x16 v; } bp, ap;
#pragma unroll
        for (int j = 0; j < 8; j++) {
          int k = ((j >> 2) << 4) + ((j & 3) << 1) + (hi << 3);   // B[k,n] = src[rbase+n][cbase+k], pair contiguous
          bp.u[j] = *(const unsigned*)&src[(rbase + nrow) * PAD + cbase + k];
          ap.u[j] = a_pack[j];
        }
        v8f acc = {};
        acc = __builtin_amdgcn_wmma_f32_16x16x32_bf16(
            false, ap.v, false, bp.v, (short)0, acc, false, false);
#pragma unroll
        for (int v = 0; v < 8; v++) {
          int colOff = v + (hi << 3);     // M = output column offset
          s_ht[(cbase + colOff) * PAD + (rbase + nrow)] = f2bf(acc[v]);
        }
      }
      __syncthreads();

      // vertical pass: 16 output fragments, one per wave; same banded A
      {
        int rbase = (wvu >> 2) * 16, cbase = (wvu & 3) * 16;
        union { unsigned u[8]; bf16x16 v; } bp, ap;
#pragma unroll
        for (int j = 0; j < 8; j++) {
          int k = ((j >> 2) << 4) + ((j & 3) << 1) + (hi << 3);   // B[k,n] = H[rbase+k][cbase+n]
          bp.u[j] = *(const unsigned*)&s_ht[(cbase + nrow) * PAD + rbase + k];
          ap.u[j] = a_pack[j];
        }
        v8f acc = {};
        acc = __builtin_amdgcn_wmma_f32_16x16x32_bf16(
            false, ap.v, false, bp.v, (short)0, acc, false, false);
#pragma unroll
        for (int v = 0; v < 8; v++) stat[m][v] = acc[v];
      }
      __syncthreads();
    }

    // per-pixel SSIM factors for this channel (all f32)
#pragma unroll
    for (int v = 0; v < 8; v++) {
      float mux = stat[0][v], muy = stat[1][v];
      float sx2 = stat[2][v] - mux * mux;
      float sy2 = stat[3][v] - muy * muy;
      float sxy = stat[4][v] - mux * muy;
      prodcs[v] *= (2.f * sxy + C2) / (sx2 + sy2 + C2);
      if (c == NC - 1)
        lterm[v] = (2.f * mux * muy + C1) / (mux * mux + muy * muy + C1);
    }
  }

  // masked tile sum -> wave reduce -> global f32 atomic
  int rg = wvu >> 2, cg = wvu & 3;
  float part = 0.f;
#pragma unroll
  for (int v = 0; v < 8; v++) {
    int orow = th0 + rg * 16 + v + (hi << 3);
    int ocol = tw0 + cg * 16 + nrow;
    if (orow < HOUT && ocol < HOUT) part += lterm[v] * prodcs[v];
  }
#pragma unroll
  for (int off = 16; off > 0; off >>= 1) part += __shfl_down(part, (unsigned)off, 32);
  if (lane == 0) atomicAdd((float*)&ws[2], part);
}

// ---------------- pass 3: finalize ----------------
__global__ void k_final(const unsigned int* __restrict__ ws, float* __restrict__ out) {
  if (threadIdx.x == 0)
    out[0] = 1.0f - ((const float*)ws)[2] / ((float)NB * (float)HOUT * (float)HOUT);
}

extern "C" void kernel_launch(void* const* d_in, const int* in_sizes, int n_in,
                              void* d_out, int out_size, void* d_ws, size_t ws_size,
                              hipStream_t stream) {
  const float* input  = (const float*)d_in[0];
  const float* target = (const float*)d_in[1];
  const float* filt   = (const float*)d_in[2];
  unsigned int* ws    = (unsigned int*)d_ws;
  float* out          = (float*)d_out;
  const int n = in_sizes[0];

  hipLaunchKernelGGL(k_init, dim3(1), dim3(32), 0, stream, ws);
  hipLaunchKernelGGL(k_minmax, dim3(2048), dim3(256), 0, stream, input, n, ws);

  const size_t lds_bytes = (size_t)(NMAPS + 1) * PAD * PAD * sizeof(unsigned short); // 76800
  hipLaunchKernelGGL(msssim_main, dim3(NB * 64), dim3(512), lds_bytes, stream,
                     input, target, filt, ws);

  hipLaunchKernelGGL(k_final, dim3(1), dim3(32), 0, stream, ws, out);
}
